// MixedModel_62878321214207
// MI455X (gfx1250) — matrix operand order repaired and checked by reading, not compiled, not run on hardware
//
#include <hip/hip_runtime.h>

typedef __attribute__((ext_vector_type(16))) _Float16 v16h;
typedef __attribute__((ext_vector_type(8)))  float    v8f;
typedef unsigned int v4u __attribute__((ext_vector_type(4)));
typedef int          v4i __attribute__((ext_vector_type(4)));
typedef int          v8i __attribute__((ext_vector_type(8)));

#define B_    16
#define S_    1024
#define D_    256
#define NB_   (B_ * S_)     // 16384 nodes
#define E_    32768
#define DH_   128
#define NC_   10

// ---------------------------------------------------------------------------
// fragment loaders (two 16B LDS loads each)
// ---------------------------------------------------------------------------
static __device__ __forceinline__ v16h ld_frag2(const _Float16* p0, const _Float16* p1) {
    union { v16h h; float4 f[2]; } u;
    u.f[0] = *(const float4*)p0;
    u.f[1] = *(const float4*)p1;
    return u.h;
}

// ---------------------------------------------------------------------------
// TDM: DMA a (rows x 32) f16 tile (row stride = rowstride_elems) into LDS.
// D# group0: count=1 | lds_addr | 57-bit global addr | type=2
// D# group1: data_size=2B, tensor dims = tile dims (tile fully in-bounds),
//            tile_dim0=32, tile_dim1=rows, tensor_dim0_stride=rowstride.
// ---------------------------------------------------------------------------
typedef __attribute__((address_space(3))) const void* lds_cptr_t;
static __device__ __forceinline__ unsigned lds_addr_of(const void* p) {
    return (unsigned)(uintptr_t)(lds_cptr_t)p;
}

static __device__ __forceinline__ void tdm_load_tile_f16(
    const _Float16* gsrc, unsigned lds_byte_off, unsigned rows, unsigned rowstride_elems)
{
    unsigned long long ga = (unsigned long long)(uintptr_t)gsrc;
    v4u g0;
    g0.x = 1u;                                   // count=1, user mode
    g0.y = lds_byte_off;                         // lds_addr
    g0.z = (unsigned)ga;                         // global_addr[31:0]
    g0.w = (unsigned)(ga >> 32) | (2u << 30);    // global_addr[56:32] | type=2
    v8i g1;
    g1[0] = (int)(1u << 16);                     // wg_mask=0, data_size=1 (2 bytes)
    g1[1] = (int)(32u << 16);                    // barrier_addr=0, tensor_dim0 lo = 32
    g1[2] = (int)(rows << 16);                   // tensor_dim0 hi=0, tensor_dim1 lo = rows
    g1[3] = (int)(32u << 16);                    // tensor_dim1 hi=0, tile_dim0 = 32
    g1[4] = (int)rows;                           // tile_dim1 = rows, tile_dim2 = 0
    g1[5] = (int)rowstride_elems;                // tensor_dim0_stride[31:0]
    g1[6] = 0;                                   // stride0 hi, stride1 lo
    g1[7] = 0;                                   // stride1 hi
    v4i z4 = { 0, 0, 0, 0 };
#if defined(__clang_major__) && (__clang_major__ >= 23)
    v8i z8 = { 0, 0, 0, 0, 0, 0, 0, 0 };
    __builtin_amdgcn_tensor_load_to_lds(g0, g1, z4, z4, z8, 0);
#else
    __builtin_amdgcn_tensor_load_to_lds(g0, g1, z4, z4, 0);
#endif
}

// ---------------------------------------------------------------------------
// Generic WMMA GEMM:  C[M,N] = A[M,K] (f16 row-major) * W[N,K]^T + bias
// block = 128 threads (4 waves), block tile 64x64, wave tile 16x64.
// K loop double-buffered: wave 0 issues TDM tile DMAs for step i+1, waits
// tensorcnt<=2 for step i, barrier republishes, all 4 waves run WMMA.
// ---------------------------------------------------------------------------
__global__ __launch_bounds__(128) void gemm_wmma(
    const _Float16* __restrict__ A, const _Float16* __restrict__ W,
    const float* __restrict__ bias, float* __restrict__ Cf,
    _Float16* __restrict__ Ch, int M, int N, int K, int relu)
{
    __shared__ _Float16 As[2][64][32];
    __shared__ _Float16 Ws[2][64][32];
    const int tid  = threadIdx.x;
    const int w    = tid >> 5;
    const int lane = tid & 31;
    const int m0   = blockIdx.x * 64;
    const int n0   = blockIdx.y * 64;
    const int nk   = K >> 5;
    const bool issuer = (w == 0);               // wave-uniform

    const unsigned ldsA0 = lds_addr_of(&As[0][0][0]);
    const unsigned ldsA1 = lds_addr_of(&As[1][0][0]);
    const unsigned ldsW0 = lds_addr_of(&Ws[0][0][0]);
    const unsigned ldsW1 = lds_addr_of(&Ws[1][0][0]);

    if (issuer) {
        tdm_load_tile_f16(A + (size_t)m0 * K, ldsA0, 64, K);
        tdm_load_tile_f16(W + (size_t)n0 * K, ldsW0, 64, K);
    }

    v8f z = {};
    v8f acc[4] = { z, z, z, z };
    const int mrow = lane & 15;
    const int kbA  = (lane < 16) ? 0 : 16;
    const int kbB  = (lane < 16) ? 0 : 16;

    for (int i = 0; i < nk; ++i) {
        if (issuer) {
            if (i + 1 < nk) {
                const int kn = (i + 1) * 32;
                tdm_load_tile_f16(A + (size_t)m0 * K + kn, ((i + 1) & 1) ? ldsA1 : ldsA0, 64, K);
                tdm_load_tile_f16(W + (size_t)n0 * K + kn, ((i + 1) & 1) ? ldsW1 : ldsW0, 64, K);
                __builtin_amdgcn_s_wait_tensorcnt((short)2);   // tile i landed (in-order)
            } else {
                __builtin_amdgcn_s_wait_tensorcnt((short)0);
            }
        }
        __syncthreads();                         // publish tile i to all waves
        const int buf = i & 1;
        const _Float16* arow = &As[buf][16 * w + mrow][0];
        v16h a = ld_frag2(arow + kbA, arow + kbA + 8);
#pragma unroll
        for (int t = 0; t < 4; ++t) {
            const _Float16* bp = &Ws[buf][16 * t + mrow][kbB];
            v16h bfrag = ld_frag2(bp, bp + 8);
            acc[t] = __builtin_amdgcn_wmma_f32_16x16x32_f16(
                false, a, false, bfrag, (short)0, acc[t], false, false);
        }
        __syncthreads();                         // all reads of buf done before reuse
    }

    const int lh = lane >> 4;
#pragma unroll
    for (int t = 0; t < 4; ++t) {
        const int col = n0 + 16 * t + mrow;
        const float bv = bias ? bias[col] : 0.0f;
#pragma unroll
        for (int r = 0; r < 8; ++r) {
            const int row = m0 + 16 * w + r + 8 * lh;
            float v = acc[t][r] + bv;
            if (relu) v = fmaxf(v, 0.0f);
            if (Cf) Cf[(size_t)row * N + col] = v;
            if (Ch) Ch[(size_t)row * N + col] = (_Float16)v;
        }
    }
}

// ---------------------------------------------------------------------------
// Flash attention: one wave per (16-query tile, batch, head). S=1024, DH=128.
// Online softmax; never materializes the S x S score matrix.
// ---------------------------------------------------------------------------
__global__ __launch_bounds__(32) void flash_attn(const _Float16* __restrict__ qkv,
                                                 _Float16* __restrict__ ao)
{
    __shared__ _Float16 Qs[16][128];
    __shared__ _Float16 Ks[32][128];
    __shared__ _Float16 Vs[32][128];
    __shared__ _Float16 Ps[16][32];

    const int lane = threadIdx.x;
    const int qt   = blockIdx.x;          // query tile (64)
    const int bh   = blockIdx.y;          // B*NHEAD (32)
    const int b    = bh >> 1, head = bh & 1;
    const size_t rowstride = 3 * D_;      // 768 halves
    const size_t base = (size_t)b * S_ * rowstride;
    const int qoff = head * DH_;
    const int koff = D_ + head * DH_;
    const int voff = 2 * D_ + head * DH_;

    // stage Q tile 16x128
    {
        int r = lane >> 1, h = (lane & 1) * 64;
        const float4* src = (const float4*)(qkv + base + (size_t)(qt * 16 + r) * rowstride + qoff + h);
        float4* dst = (float4*)(&Qs[r][h]);
#pragma unroll
        for (int i = 0; i < 8; ++i) dst[i] = src[i];
    }
    __syncthreads();

    const int mrow = lane & 15;
    const int kbA  = (lane < 16) ? 0 : 16;
    const int kbB  = (lane < 16) ? 0 : 16;
    const int lh   = lane >> 4;

    v16h qf[4];
#pragma unroll
    for (int s = 0; s < 4; ++s) {
        const _Float16* p = &Qs[mrow][32 * s + kbA];
        qf[s] = ld_frag2(p, p + 8);
    }

    v8f z = {};
    v8f o[8] = { z, z, z, z, z, z, z, z };
    float m_row[8], l_row[8];
#pragma unroll
    for (int r = 0; r < 8; ++r) { m_row[r] = -1e30f; l_row[r] = 0.0f; }
    const float scale = 0.08838834764831845f;   // 1/sqrt(128)

    for (int j0 = 0; j0 < S_; j0 += 32) {
        {   // stage K,V tiles 32x128
            const float4* srcK = (const float4*)(qkv + base + (size_t)(j0 + lane) * rowstride + koff);
            const float4* srcV = (const float4*)(qkv + base + (size_t)(j0 + lane) * rowstride + voff);
            float4* dstK = (float4*)(&Ks[lane][0]);
            float4* dstV = (float4*)(&Vs[lane][0]);
#pragma unroll
            for (int i = 0; i < 16; ++i) { dstK[i] = srcK[i]; dstV[i] = srcV[i]; }
        }
        __syncthreads();

        // S tile = Q (16x128) * K^T  -> two 16x16 col-tiles, K dim in 4 slices
        v8f s0 = z, s1 = z;
#pragma unroll
        for (int s = 0; s < 4; ++s) {
            const _Float16* p0 = &Ks[mrow][32 * s + kbB];
            v16h k0 = ld_frag2(p0, p0 + 8);
            const _Float16* p1 = &Ks[16 + mrow][32 * s + kbB];
            v16h k1 = ld_frag2(p1, p1 + 8);
            s0 = __builtin_amdgcn_wmma_f32_16x16x32_f16(false, qf[s], false, k0, (short)0, s0, false, false);
            s1 = __builtin_amdgcn_wmma_f32_16x16x32_f16(false, qf[s], false, k1, (short)0, s1, false, false);
        }

        // online softmax over 32 new columns (rows live across 16-lane halves)
#pragma unroll
        for (int r = 0; r < 8; ++r) {
            float a0 = s0[r] * scale, a1 = s1[r] * scale;
            float v = fmaxf(a0, a1);
#pragma unroll
            for (int msk = 1; msk < 16; msk <<= 1) v = fmaxf(v, __shfl_xor(v, msk));
            float mn = fmaxf(m_row[r], v);
            float p0 = __expf(a0 - mn), p1 = __expf(a1 - mn);
            float ps = p0 + p1;
#pragma unroll
            for (int msk = 1; msk < 16; msk <<= 1) ps += __shfl_xor(ps, msk);
            float alpha = __expf(m_row[r] - mn);
            l_row[r] = l_row[r] * alpha + ps;
            m_row[r] = mn;
#pragma unroll
            for (int t = 0; t < 8; ++t) o[t][r] *= alpha;
            Ps[r + 8 * lh][mrow]      = (_Float16)p0;
            Ps[r + 8 * lh][16 + mrow] = (_Float16)p1;
        }
        __syncthreads();

        // O += P (16x32) * V (32x128)
        const _Float16* pp = &Ps[mrow][kbA];
        v16h pf = ld_frag2(pp, pp + 8);
#pragma unroll
        for (int t = 0; t < 8; ++t) {
            v16h vf;
#pragma unroll
            for (int e = 0; e < 16; ++e) vf[e] = Vs[kbB + e][16 * t + mrow];
            o[t] = __builtin_amdgcn_wmma_f32_16x16x32_f16(false, pf, false, vf, (short)0, o[t], false, false);
        }
        __syncthreads();
    }

    float inv[8];
#pragma unroll
    for (int r = 0; r < 8; ++r) inv[r] = 1.0f / l_row[r];
#pragma unroll
    for (int t = 0; t < 8; ++t) {
#pragma unroll
        for (int r = 0; r < 8; ++r) {
            int srow = qt * 16 + r + 8 * lh;
            int col  = head * DH_ + 16 * t + mrow;
            ao[((size_t)b * S_ + srow) * D_ + col] = (_Float16)(o[t][r] * inv[r]);
        }
    }
}

// ---------------------------------------------------------------------------
// elementwise / LN / embed / GCN / pooling kernels
// ---------------------------------------------------------------------------
__global__ __launch_bounds__(256) void embed_kernel(const int* __restrict__ x,
    const float* __restrict__ emb, float* __restrict__ hf, _Float16* __restrict__ hh)
{
    const int i = blockIdx.x, d = threadIdx.x;
    const int s = i & (S_ - 1);
    const int tok = x[i];
    const int i2 = d & ~1;
    const float freq = __expf(-(float)i2 * (9.210340371976184f / 256.0f));
    const float ang  = (float)s * freq;
    const float pe   = (d & 1) ? __cosf(ang) : __sinf(ang);
    float v = emb[(size_t)tok * 256 + d] * 16.0f + pe;   // sqrt(D)=16
    hf[(size_t)i * 256 + d] = v;
    hh[(size_t)i * 256 + d] = (_Float16)v;
}

__global__ __launch_bounds__(256) void add_ln(const float* __restrict__ X,
    const float* __restrict__ Y, const float* __restrict__ sc, const float* __restrict__ bi,
    float* __restrict__ Of, _Float16* __restrict__ Oh)
{
    __shared__ float red[256];
    const int i = blockIdx.x, d = threadIdx.x;
    float v = X[(size_t)i * 256 + d] + Y[(size_t)i * 256 + d];
    red[d] = v; __syncthreads();
    for (int st = 128; st > 0; st >>= 1) { if (d < st) red[d] += red[d + st]; __syncthreads(); }
    float mean = red[0] * (1.0f / 256.0f); __syncthreads();
    float c = v - mean;
    red[d] = c * c; __syncthreads();
    for (int st = 128; st > 0; st >>= 1) { if (d < st) red[d] += red[d + st]; __syncthreads(); }
    float var = red[0] * (1.0f / 256.0f);
    float o = c * rsqrtf(var + 1e-5f) * sc[d] + bi[d];
    Of[(size_t)i * 256 + d] = o;
    Oh[(size_t)i * 256 + d] = (_Float16)o;
}

__global__ void cvt_f16(const float* __restrict__ src, _Float16* __restrict__ dst, int n) {
    int i = blockIdx.x * 256 + threadIdx.x;
    if (i < n) dst[i] = (_Float16)src[i];
}
__global__ void cvt_f16_T(const float* __restrict__ src, _Float16* __restrict__ dst) {
    int i = blockIdx.x * 256 + threadIdx.x;          // 65536 elems, [256,256]
    int n = i >> 8, k = i & 255;
    dst[(size_t)n * 256 + k] = (_Float16)src[(size_t)k * 256 + n];
}

__global__ void deg_init(float* deg) { deg[blockIdx.x * 256 + threadIdx.x] = 1.0f; }

__global__ void deg_edges(const int* __restrict__ ei, float* __restrict__ deg) {
    int idx = blockIdx.x * 256 + threadIdx.x;        // B*E
    int b = idx >> 15, le = idx & (E_ - 1);
    int col = ei[E_ + le] + b * S_;
    atomicAdd(&deg[col], 1.0f);
}
__global__ void dinv_k(const float* __restrict__ deg, float* __restrict__ dinv) {
    int i = blockIdx.x * 256 + threadIdx.x;
    dinv[i] = rsqrtf(deg[i]);
}
__global__ __launch_bounds__(256) void gcn_init(const float* __restrict__ xw,
    const float* __restrict__ dinv, const float* __restrict__ bias, float* __restrict__ gout)
{
    int i = blockIdx.x, c = threadIdx.x;
    float di = dinv[i];
    gout[(size_t)i * 256 + c] = bias[c] + xw[(size_t)i * 256 + c] * di * di;
}
__global__ __launch_bounds__(256) void gcn_scatter(const int* __restrict__ ei,
    const float* __restrict__ xw, const float* __restrict__ dinv, float* __restrict__ gout)
{
    int e = blockIdx.x * 8 + (threadIdx.x >> 5);     // one wave per edge
    int lane = threadIdx.x & 31;
    int b = e >> 15, le = e & (E_ - 1);
    int row = ei[le] + b * S_;
    int col = ei[E_ + le] + b * S_;
    float norm = dinv[row] * dinv[col];
#pragma unroll
    for (int j = 0; j < 8; ++j) {
        int c = lane * 8 + j;
        atomicAdd(&gout[(size_t)col * 256 + c], xw[(size_t)row * 256 + c] * norm);
    }
}

__global__ __launch_bounds__(128) void attn_pool_score(const float* __restrict__ gout,
    const float* __restrict__ w1, const float* __restrict__ b1,
    const float* __restrict__ w2, const float* __restrict__ b2, float* __restrict__ apre)
{
    __shared__ float red[128];
    int i = blockIdx.x, j = threadIdx.x;
    float acc = 0.f;
    for (int k = 0; k < 256; ++k) acc += w1[j * 256 + k] * gout[(size_t)i * 256 + k];
    red[j] = tanhf(acc + b1[j]) * w2[j];
    __syncthreads();
    for (int st = 64; st > 0; st >>= 1) { if (j < st) red[j] += red[j + st]; __syncthreads(); }
    if (j == 0) apre[i] = red[0] + b2[0];
}

__global__ __launch_bounds__(1024) void softmax_all(const float* __restrict__ a,
                                                    float* __restrict__ an)
{
    __shared__ float red[1024];
    __shared__ float smax, ssum;
    int t = threadIdx.x;
    float m = -1e30f;
    for (int i = t; i < NB_; i += 1024) m = fmaxf(m, a[i]);
    red[t] = m; __syncthreads();
    for (int st = 512; st > 0; st >>= 1) { if (t < st) red[t] = fmaxf(red[t], red[t + st]); __syncthreads(); }
    if (t == 0) smax = red[0];
    __syncthreads();
    float s = 0.f;
    for (int i = t; i < NB_; i += 1024) s += __expf(a[i] - smax);
    red[t] = s; __syncthreads();
    for (int st = 512; st > 0; st >>= 1) { if (t < st) red[t] += red[t + st]; __syncthreads(); }
    if (t == 0) ssum = red[0];
    __syncthreads();
    float inv = 1.0f / ssum;
    for (int i = t; i < NB_; i += 1024) an[i] = __expf(a[i] - smax) * inv;
}

__global__ __launch_bounds__(256) void pool_max(const float* __restrict__ gout,
    const float* __restrict__ an, float* __restrict__ pooled)
{
    int b = blockIdx.x, c = threadIdx.x;
    float m = -1e30f;
    for (int s = 0; s < S_; ++s) {
        size_t i = (size_t)b * S_ + s;
        m = fmaxf(m, gout[i * 256 + c] * an[i]);
    }
    pooled[b * 256 + c] = m;
}

__global__ __launch_bounds__(128) void classifier(const float* __restrict__ pooled,
    const float* __restrict__ w1, const float* __restrict__ b1,
    const float* __restrict__ w2, const float* __restrict__ b2, float* __restrict__ out)
{
    __shared__ float h1[128];
    int b = blockIdx.x, j = threadIdx.x;
    float acc = 0.f;
    for (int k = 0; k < 256; ++k) acc += w1[j * 256 + k] * pooled[b * 256 + k];
    h1[j] = fmaxf(acc + b1[j], 0.0f);
    __syncthreads();
    if (j < NC_) {
        float s = 0.f;
        for (int k = 0; k < 128; ++k) s += w2[j * 128 + k] * h1[k];
        out[b * NC_ + j] = s + b2[j];
    }
}

// ---------------------------------------------------------------------------
// launcher
// ---------------------------------------------------------------------------
extern "C" void kernel_launch(void* const* d_in, const int* in_sizes, int n_in,
                              void* d_out, int out_size, void* d_ws, size_t ws_size,
                              hipStream_t stream)
{
    (void)in_sizes; (void)n_in; (void)out_size; (void)ws_size;
    const int*   x     = (const int*)d_in[0];
    const int*   ei    = (const int*)d_in[1];
    const float* emb   = (const float*)d_in[2];
    const float* in_w  = (const float*)d_in[3];
    const float* in_b  = (const float*)d_in[4];
    const float* out_w = (const float*)d_in[5];
    const float* out_b = (const float*)d_in[6];
    const float* ln1s  = (const float*)d_in[7];
    const float* ln1b  = (const float*)d_in[8];
    const float* ln2s  = (const float*)d_in[9];
    const float* ln2b  = (const float*)d_in[10];
    const float* f1w   = (const float*)d_in[11];
    const float* f1b   = (const float*)d_in[12];
    const float* f2w   = (const float*)d_in[13];
    const float* f2b   = (const float*)d_in[14];
    const float* gw    = (const float*)d_in[15];
    const float* gb    = (const float*)d_in[16];
    const float* aw1   = (const float*)d_in[17];
    const float* ab1   = (const float*)d_in[18];
    const float* aw2   = (const float*)d_in[19];
    const float* ab2   = (const float*)d_in[20];
    const float* cw1   = (const float*)d_in[21];
    const float* cb1   = (const float*)d_in[22];
    const float* cw2   = (const float*)d_in[23];
    const float* cb2   = (const float*)d_in[24];

    char* ws = (char*)d_ws;
    size_t off = 0;
    auto alloc = [&](size_t bytes) -> void* {
        void* p = ws + off;
        off += (bytes + 255) & ~(size_t)255;
        return p;
    };
    _Float16* Win   = (_Float16*)alloc((size_t)768 * 256 * 2);
    _Float16* Wout  = (_Float16*)alloc((size_t)256 * 256 * 2);
    _Float16* Wf1   = (_Float16*)alloc((size_t)512 * 256 * 2);
    _Float16* Wf2   = (_Float16*)alloc((size_t)256 * 512 * 2);
    _Float16* Wg    = (_Float16*)alloc((size_t)256 * 256 * 2);
    float*    hf32  = (float*)   alloc((size_t)NB_ * 256 * 4);
    _Float16* hf16  = (_Float16*)alloc((size_t)NB_ * 256 * 2);
    _Float16* qkv16 = (_Float16*)alloc((size_t)NB_ * 768 * 2);
    _Float16* ao16  = (_Float16*)alloc((size_t)NB_ * 256 * 2);
    float*    tmp32 = (float*)   alloc((size_t)NB_ * 256 * 4);  // out_proj out / ffn2 out / xw
    float*    h1f32 = (float*)   alloc((size_t)NB_ * 256 * 4);
    _Float16* h1f16 = (_Float16*)alloc((size_t)NB_ * 256 * 2);
    _Float16* ff16  = (_Float16*)alloc((size_t)NB_ * 512 * 2);
    float*    h2f32 = (float*)   alloc((size_t)NB_ * 256 * 4);
    _Float16* h2f16 = (_Float16*)alloc((size_t)NB_ * 256 * 2);
    float*    gout  = (float*)   alloc((size_t)NB_ * 256 * 4);
    float*    deg   = (float*)   alloc((size_t)NB_ * 4);
    float*    dinv  = (float*)   alloc((size_t)NB_ * 4);
    float*    apre  = (float*)   alloc((size_t)NB_ * 4);
    float*    anorm = (float*)   alloc((size_t)NB_ * 4);
    float*    pooled= (float*)   alloc((size_t)B_ * 256 * 4);

    // weight precision conversion (f32 -> f16; gcn_w also transposed to [H,D])
    cvt_f16<<<768, 256, 0, stream>>>(in_w,  Win,  768 * 256);
    cvt_f16<<<256, 256, 0, stream>>>(out_w, Wout, 256 * 256);
    cvt_f16<<<512, 256, 0, stream>>>(f1w,   Wf1,  512 * 256);
    cvt_f16<<<512, 256, 0, stream>>>(f2w,   Wf2,  256 * 512);
    cvt_f16_T<<<256, 256, 0, stream>>>(gw, Wg);

    // transformer layer
    embed_kernel<<<NB_, 256, 0, stream>>>(x, emb, hf32, hf16);
    gemm_wmma<<<dim3(NB_ / 64, 768 / 64), 128, 0, stream>>>(hf16, Win, in_b, nullptr, qkv16, NB_, 768, 256, 0);
    flash_attn<<<dim3(S_ / 16, B_ * 2), 32, 0, stream>>>(qkv16, ao16);
    gemm_wmma<<<dim3(NB_ / 64, 256 / 64), 128, 0, stream>>>(ao16, Wout, out_b, tmp32, nullptr, NB_, 256, 256, 0);
    add_ln<<<NB_, 256, 0, stream>>>(hf32, tmp32, ln1s, ln1b, h1f32, h1f16);
    gemm_wmma<<<dim3(NB_ / 64, 512 / 64), 128, 0, stream>>>(h1f16, Wf1, f1b, nullptr, ff16, NB_, 512, 256, 1);
    gemm_wmma<<<dim3(NB_ / 64, 256 / 64), 128, 0, stream>>>(ff16, Wf2, f2b, tmp32, nullptr, NB_, 256, 512, 0);
    add_ln<<<NB_, 256, 0, stream>>>(h1f32, tmp32, ln2s, ln2b, h2f32, h2f16);

    // GCN: xw = nodes @ gcn_w (via transposed f16 weight), then normalized scatter
    gemm_wmma<<<dim3(NB_ / 64, 256 / 64), 128, 0, stream>>>(h2f16, Wg, nullptr, tmp32, nullptr, NB_, 256, 256, 0);
    deg_init<<<NB_ / 256, 256, 0, stream>>>(deg);
    deg_edges<<<(B_ * E_) / 256, 256, 0, stream>>>(ei, deg);
    dinv_k<<<NB_ / 256, 256, 0, stream>>>(deg, dinv);
    gcn_init<<<NB_, 256, 0, stream>>>(tmp32, dinv, gb, gout);
    gcn_scatter<<<(B_ * E_) / 8, 256, 0, stream>>>(ei, tmp32, dinv, gout);

    // attention pooling over all nodes + classifier
    attn_pool_score<<<NB_, 128, 0, stream>>>(gout, aw1, ab1, aw2, ab2, apre);
    softmax_all<<<1, 1024, 0, stream>>>(apre, anorm);
    pool_max<<<B_, 256, 0, stream>>>(gout, anorm, pooled);
    classifier<<<B_, 128, 0, stream>>>(pooled, cw1, cb1, cw2, cb2, (float*)d_out);
}
